// Sum_40278203302486
// MI455X (gfx1250) — matrix operand order repaired and checked by reading, not compiled, not run on hardware
//
#include <hip/hip_runtime.h>
#include <hip/hip_bf16.h>

typedef __attribute__((ext_vector_type(16))) _Float16 v16h;
typedef __attribute__((ext_vector_type(8)))  _Float16 v8h;
typedef __attribute__((ext_vector_type(8)))  float    v8f;

#define NN  128
#define DD  512
#define ICc 32
#define OCc 8
#define RRc 4
#define ICR (ICc*RRc)     // 128 contiguous floats per (n,d) in x/vars
#define NH  64            // rows (n) handled per workgroup
#define APAD 40           // 32 ic halves + 8 pad halves (16B) -> conflict-free b128

__global__ __launch_bounds__(256, 2)
void gmm_lse_kernel(const float* __restrict__ x,
                    const float* __restrict__ vars,
                    const float* __restrict__ w,
                    float* __restrict__ out_probs,
                    float* __restrict__ out_vars)
{
    const int d    = blockIdx.x >> 1;          // which D slice
    const int nblk = (blockIdx.x & 1) * NH;    // which half of N
    const int t    = threadIdx.x;

    // A-operand staging: exp(x-mx), exp(v-mv) in WMMA 16-bit A layout feed
    __shared__ alignas(16) _Float16 EA[RRc][NH][APAD];
    __shared__ alignas(16) _Float16 VA[RRc][NH][APAD];
    // B operands: [0]=wn, [1]=wn^2 ; [r][col(16)][ic(32)+pad]
    __shared__ alignas(16) _Float16 BW[2][RRc][16][APAD];
    __shared__ float MX[NH * RRc];
    __shared__ float MV[NH * RRc];
    __shared__ float WRAW[ICc * OCc * RRc];    // 1024 floats = this d's weight slice

    // ---- 1) load weight slice (contiguous 4KB), coalesced float4 ----
    const float4* wsrc = (const float4*)(w + (size_t)d * (ICc * OCc * RRc));
    ((float4*)WRAW)[t] = wsrc[t];
    __syncthreads();

    // ---- 2) softmax over IC -> wn, wn^2 into B-fragment layout ----
    if (t < 32) {
        // column id in raw slice: ic*32 + (oc*4 + r); t == oc*4+r
        const int oc = t >> 2, r = t & 3;
        float m = -INFINITY;
        #pragma unroll
        for (int ic = 0; ic < ICc; ++ic) m = fmaxf(m, WRAW[ic * 32 + t]);
        float s = 0.f;
        #pragma unroll
        for (int ic = 0; ic < ICc; ++ic) s += __expf(WRAW[ic * 32 + t] - m);
        const float inv = 1.0f / s;
        #pragma unroll
        for (int ic = 0; ic < ICc; ++ic) {
            float wn = __expf(WRAW[ic * 32 + t] - m) * inv;
            BW[0][r][oc][ic] = (_Float16)wn;
            BW[1][r][oc][ic] = (_Float16)(wn * wn);
        }
    } else if (t < 64) {
        const int tt = t - 32;
        const int oc = 8 + (tt >> 2), r = tt & 3;   // zero unused B columns 8..15
        #pragma unroll
        for (int ic = 0; ic < ICc; ++ic) {
            BW[0][r][oc][ic] = (_Float16)0.f;
            BW[1][r][oc][ic] = (_Float16)0.f;
        }
    }

    // ---- 3) per-(n,r) row: load 32 ics, rowmax, exp -> f16 LDS ----
    {
        const int nl = t >> 2;             // local n: 0..63
        const int r  = t & 3;
        const int n  = nblk + nl;
        const size_t base = ((size_t)n * DD + d) * ICR + r;

        float vreg[ICc];
        const float* xp = x + base;
        #pragma unroll
        for (int ic = 0; ic < ICc; ++ic) vreg[ic] = xp[ic * 4];
        float m = -INFINITY;
        #pragma unroll
        for (int ic = 0; ic < ICc; ++ic) m = fmaxf(m, vreg[ic]);
        MX[nl * 4 + r] = m;
        #pragma unroll
        for (int ic = 0; ic < ICc; ++ic)
            EA[r][nl][ic] = (_Float16)__expf(vreg[ic] - m);

        const float* vp = vars + base;
        #pragma unroll
        for (int ic = 0; ic < ICc; ++ic) vreg[ic] = vp[ic * 4];
        m = -INFINITY;
        #pragma unroll
        for (int ic = 0; ic < ICc; ++ic) m = fmaxf(m, vreg[ic]);
        MV[nl * 4 + r] = m;
        #pragma unroll
        for (int ic = 0; ic < ICc; ++ic)
            VA[r][nl][ic] = (_Float16)__expf(vreg[ic] - m);
    }
    __syncthreads();

    // ---- 4) WMMA: 4 n-tiles x 4 r = 16 (tile,r) pairs over 8 waves ----
    const int wave  = t >> 5;
    const int lane  = t & 31;
    const int lrow  = lane & 15;   // A row / B,D column within tile
    const int lhalf = lane >> 4;   // half-wave select

    for (int p = wave; p < 16; p += 8) {
        const int tile = p >> 2, r = p & 3;
        const int n0   = tile * 16;

        // A fragments: lane half h: K = h%8 + 16*(h/8) + 8*lhalf, row = n0+lrow
        v8h aLo = *(const v8h*)&EA[r][n0 + lrow][8 * lhalf];
        v8h aHi = *(const v8h*)&EA[r][n0 + lrow][16 + 8 * lhalf];
        v16h aE = __builtin_shufflevector(aLo, aHi, 0,1,2,3,4,5,6,7,8,9,10,11,12,13,14,15);
        v8h vLo = *(const v8h*)&VA[r][n0 + lrow][8 * lhalf];
        v8h vHi = *(const v8h*)&VA[r][n0 + lrow][16 + 8 * lhalf];
        v16h aV = __builtin_shufflevector(vLo, vHi, 0,1,2,3,4,5,6,7,8,9,10,11,12,13,14,15);
        v16h aE2 = aE * aE;                         // exp(2(x-mx)) as packed-f16 square

        // B fragments: lane half h: K = h%8 + 16*(h/8) + 8*lhalf, col = lrow
        v8h bLo  = *(const v8h*)&BW[0][r][lrow][8 * lhalf];
        v8h bHi  = *(const v8h*)&BW[0][r][lrow][16 + 8 * lhalf];
        v16h bW  = __builtin_shufflevector(bLo, bHi, 0,1,2,3,4,5,6,7,8,9,10,11,12,13,14,15);
        v8h b2Lo = *(const v8h*)&BW[1][r][lrow][8 * lhalf];
        v8h b2Hi = *(const v8h*)&BW[1][r][lrow][16 + 8 * lhalf];
        v16h bW2 = __builtin_shufflevector(b2Lo, b2Hi, 0,1,2,3,4,5,6,7,8,9,10,11,12,13,14,15);

        v8f c0 = {};  // sum wn   * exp(x-mx)
        v8f c1 = {};  // sum wn^2 * exp(2(x-mx))
        v8f c2 = {};  // sum wn^2 * exp(v-mv)
        c0 = __builtin_amdgcn_wmma_f32_16x16x32_f16(false, aE,  false, bW,  (short)0, c0, false, false);
        c1 = __builtin_amdgcn_wmma_f32_16x16x32_f16(false, aE2, false, bW2, (short)0, c1, false, false);
        c2 = __builtin_amdgcn_wmma_f32_16x16x32_f16(false, aV,  false, bW2, (short)0, c2, false, false);

        // ---- 5) epilogue: D layout lane holds col=lrow, rows M=v+8*lhalf ----
        if (lrow < OCc) {
            const int oc = lrow;
            #pragma unroll
            for (int v = 0; v < 8; ++v) {
                const int nl  = n0 + v + 8 * lhalf;     // local n in [0,64)
                const int rw  = nl * 4 + r;
                const int n   = nblk + nl;
                const float mx = MX[rw];
                const float mv = MV[rw];
                const size_t o = ((size_t)n * DD + d) * (OCc * RRc) + oc * RRc + r;

                out_probs[o] = __logf(c0[v]) + mx;

                const float rt = 2.0f * mx + __logf(c1[v]);
                const float lt = mv + __logf(c2[v]);
                const float mm = fmaxf(lt, rt);
                out_vars[o] = mm + __logf(__expf(lt - mm) + __expf(rt - mm));
            }
        }
    }
}

extern "C" void kernel_launch(void* const* d_in, const int* in_sizes, int n_in,
                              void* d_out, int out_size, void* d_ws, size_t ws_size,
                              hipStream_t stream) {
    (void)in_sizes; (void)n_in; (void)out_size; (void)d_ws; (void)ws_size;
    const float* x    = (const float*)d_in[0];
    const float* vars = (const float*)d_in[1];
    const float* w    = (const float*)d_in[2];
    float* out_probs  = (float*)d_out;
    float* out_vars   = out_probs + (size_t)NN * DD * OCc * RRc;  // tuple concat

    gmm_lse_kernel<<<dim3(DD * 2), dim3(256), 0, stream>>>(x, vars, w, out_probs, out_vars);
}
